// VoxelNet_57397942944142
// MI455X (gfx1250) — matrix operand order repaired and checked by reading, not compiled, not run on hardware
//
#include <hip/hip_runtime.h>
#include <hip/hip_bf16.h>
#include <stdint.h>

typedef _Float16 v16h  __attribute__((ext_vector_type(16)));
typedef _Float16 half8 __attribute__((ext_vector_type(8)));
typedef float    v8f   __attribute__((ext_vector_type(8)));
typedef float    v4f   __attribute__((ext_vector_type(4)));

// ---- problem constants ----
#define DD   4
#define HH   200
#define WW   200
#define SPOS 160000      // DD*HH*WW
#define HWN  40000       // HH*WW
#define HID  64
#define OUTC 256
#define NWT  13          // ceil(200/16) W-tiles

// padded activation grid: [DD+2][HH+2][WP][C], halo of 1, W padded to 212
#define HP   202
#define WP   212

// ---- workspace offsets (bytes, all 16B aligned) ----
#define OFF_X1P  ((size_t)0)           // f16 [6][202][212][32]   16,444,416
#define OFF_X2P  ((size_t)16444416)    // f16 [6][202][212][64]   32,888,832
#define OFF_X3   ((size_t)49333248)    // f16 [40000][256]        20,480,000
#define OFF_Y    ((size_t)69813248)    // f32 [160000][64]        40,960,000
#define OFF_W1P  ((size_t)110773248)   // f16 [27][64][32]           110,592
#define OFF_W2P  ((size_t)110883840)   // f16 [27][64][64]           221,184
#define OFF_WHP  ((size_t)111105024)   // f16 [256][256]             131,072
#define OFF_PART ((size_t)111236096)   // f32 [157][128]              80,384
#define OFF_SS   ((size_t)111316480)   // f32 [128]                      512
#define ZERO_N4  3083328               // (OFF_X1P sz + OFF_X2P sz)/16 float4s

static __device__ __forceinline__ v16h combine16(half8 lo, half8 hi) {
  v16h r;
#pragma unroll
  for (int i = 0; i < 8; ++i) { r[i] = lo[i]; r[8 + i] = hi[i]; }
  return r;
}

// ---------------- setup kernels ----------------

__global__ void k_zero16(float4* __restrict__ p, int n4) {   // zero 16B chunks
  int i = blockIdx.x * blockDim.x + threadIdx.x;
  if (i < n4) {
    float4 z; z.x = z.y = z.z = z.w = 0.f;
    p[i] = z;
  }
}

// voxels[V][4][32] mean over P=32 -> scatter into padded x1 [z+1][y+1][x+1][c]
__global__ void k_voxel(const float* __restrict__ vox, const int* __restrict__ idx,
                        _Float16* __restrict__ x1) {
  int tid = blockIdx.x * blockDim.x + threadIdx.x;   // 160000 = 40000 voxels * 4 ch
  int v = tid >> 2, ci = tid & 3;
  const float* src = vox + ((size_t)v * 4 + ci) * 32;
  float s = 0.f;
#pragma unroll
  for (int i = 0; i < 32; ++i) s += src[i];
  s *= (1.0f / 32.0f);
  int xi = idx[v * 3 + 0], yi = idx[v * 3 + 1], zi = idx[v * 3 + 2];
  x1[(((size_t)(zi + 1) * HP + (yi + 1)) * WP + (xi + 1)) * 32 + ci] = (_Float16)s;
}

// w1 f32 [64][4][27] -> w1p f16 [27][64][32] (ic padded to 32 with zeros)
__global__ void k_prep_w1(const float* __restrict__ w, _Float16* __restrict__ wp) {
  int o = blockIdx.x * blockDim.x + threadIdx.x;     // 55296
  int t = o / 2048, r = o % 2048, oc = r / 32, ic = r % 32;
  float v = (ic < 4) ? w[(size_t)oc * 108 + ic * 27 + t] : 0.f;
  wp[o] = (_Float16)v;
}

// w2 f32 [64][64][27] -> w2p f16 [27][64][64]
__global__ void k_prep_w2(const float* __restrict__ w, _Float16* __restrict__ wp) {
  int o = blockIdx.x * blockDim.x + threadIdx.x;     // 110592
  int t = o / 4096, r = o % 4096, oc = r / 64, ic = r % 64;
  wp[o] = (_Float16)w[(size_t)oc * 1728 + ic * 27 + t];
}

// wh f32 [256][256] -> f16 same layout
__global__ void k_prep_wh(const float* __restrict__ w, _Float16* __restrict__ wp) {
  int o = blockIdx.x * blockDim.x + threadIdx.x;     // 65536
  wp[o] = (_Float16)w[o];
}

// ------- conv1: x1p [6][202][212][32] * w1p [27][64][32] -> y [160000][64] -------
// one wave computes 16 W-positions x all 64 out-channels (4 accumulators, B reused)
__global__ __launch_bounds__(32) void k_conv1(
    const _Float16* __restrict__ xp, const _Float16* __restrict__ wp,
    const float* __restrict__ bias, float* __restrict__ y) {
  const int lane = threadIdx.x;
  const int mn = lane & 15, lh = lane >> 4;
  const int w0 = blockIdx.x * 16;
  const int h = blockIdx.y, d = blockIdx.z;

  v8f acc[4];
#pragma unroll
  for (int mt = 0; mt < 4; ++mt) {
    const float* bp = bias + mt * 16 + 8 * lh;
    v4f b0 = *(const v4f*)bp, b1 = *(const v4f*)(bp + 4);
#pragma unroll
    for (int i = 0; i < 4; ++i) { acc[mt][i] = b0[i]; acc[mt][4 + i] = b1[i]; }
  }

#pragma unroll
  for (int dz = 0; dz < 3; ++dz) {
#pragma unroll
    for (int dy = 0; dy < 3; ++dy) {
#pragma unroll
      for (int dx = 0; dx < 3; ++dx) {
        const int t = (dz * 3 + dy) * 3 + dx;
        const _Float16* xrow = xp +
            ((size_t)(((d + dz) * HP + (h + dy)) * WP + (w0 + mn + dx))) * 32;
        v16h b = combine16(*(const half8*)(xrow + 8 * lh),
                           *(const half8*)(xrow + 16 + 8 * lh));
#pragma unroll
        for (int mt = 0; mt < 4; ++mt) {
          const _Float16* wrow = wp + ((size_t)t * 64 + mt * 16 + mn) * 32;
          v16h a = combine16(*(const half8*)(wrow + 8 * lh),
                             *(const half8*)(wrow + 16 + 8 * lh));
          acc[mt] = __builtin_amdgcn_wmma_f32_16x16x32_f16(false, a, false, b,
                                                           (short)0, acc[mt], false, false);
        }
      }
    }
  }

  const int wn = w0 + mn;
  if (wn < WW) {
    float* ybase = y + ((size_t)((d * HH + h) * WW + wn)) * 64 + 8 * lh;
#pragma unroll
    for (int mt = 0; mt < 4; ++mt) {
      v4f s0 = {acc[mt][0], acc[mt][1], acc[mt][2], acc[mt][3]};
      v4f s1 = {acc[mt][4], acc[mt][5], acc[mt][6], acc[mt][7]};
      *(v4f*)(ybase + mt * 16) = s0;
      *(v4f*)(ybase + mt * 16 + 4) = s1;
    }
  }
}

// ------- conv2: x2p [6][202][212][64] * w2p [27][64][64] -> y [160000][64] -------
__global__ __launch_bounds__(32) void k_conv2(
    const _Float16* __restrict__ xp, const _Float16* __restrict__ wp,
    const float* __restrict__ bias, float* __restrict__ y) {
  const int lane = threadIdx.x;
  const int mn = lane & 15, lh = lane >> 4;
  const int w0 = blockIdx.x * 16;
  const int h = blockIdx.y, d = blockIdx.z;

  v8f acc[4];
#pragma unroll
  for (int mt = 0; mt < 4; ++mt) {
    const float* bp = bias + mt * 16 + 8 * lh;
    v4f b0 = *(const v4f*)bp, b1 = *(const v4f*)(bp + 4);
#pragma unroll
    for (int i = 0; i < 4; ++i) { acc[mt][i] = b0[i]; acc[mt][4 + i] = b1[i]; }
  }

#pragma unroll
  for (int dz = 0; dz < 3; ++dz) {
#pragma unroll
    for (int dy = 0; dy < 3; ++dy) {
#pragma unroll
      for (int dx = 0; dx < 3; ++dx) {
        const int t = (dz * 3 + dy) * 3 + dx;
        const _Float16* xrow = xp +
            ((size_t)(((d + dz) * HP + (h + dy)) * WP + (w0 + mn + dx))) * 64;
#pragma unroll
        for (int ks = 0; ks < 2; ++ks) {
          const int k0 = ks * 32;
          v16h b = combine16(*(const half8*)(xrow + k0 + 8 * lh),
                             *(const half8*)(xrow + k0 + 16 + 8 * lh));
#pragma unroll
          for (int mt = 0; mt < 4; ++mt) {
            const _Float16* wrow = wp + ((size_t)t * 64 + mt * 16 + mn) * 64;
            v16h a = combine16(*(const half8*)(wrow + k0 + 8 * lh),
                               *(const half8*)(wrow + k0 + 16 + 8 * lh));
            acc[mt] = __builtin_amdgcn_wmma_f32_16x16x32_f16(false, a, false, b,
                                                             (short)0, acc[mt], false, false);
          }
        }
      }
    }
  }

  const int wn = w0 + mn;
  if (wn < WW) {
    float* ybase = y + ((size_t)((d * HH + h) * WW + wn)) * 64 + 8 * lh;
#pragma unroll
    for (int mt = 0; mt < 4; ++mt) {
      v4f s0 = {acc[mt][0], acc[mt][1], acc[mt][2], acc[mt][3]};
      v4f s1 = {acc[mt][4], acc[mt][5], acc[mt][6], acc[mt][7]};
      *(v4f*)(ybase + mt * 16) = s0;
      *(v4f*)(ybase + mt * 16 + 4) = s1;
    }
  }
}

// ---------------- BN stats (deterministic two-stage) ----------------
__global__ void k_bnstat(const float* __restrict__ y, float* __restrict__ part) {
  __shared__ float ssm[256];
  __shared__ float sqm[256];
  int c = threadIdx.x & 63, g = threadIdx.x >> 6;
  float s = 0.f, q = 0.f;
  int base = blockIdx.x * 1024;
  for (int i = g; i < 1024; i += 4) {
    int p = base + i;
    if (p < SPOS) {
      float v = y[(size_t)p * 64 + c];
      s += v; q += v * v;
    }
  }
  ssm[threadIdx.x] = s; sqm[threadIdx.x] = q;
  __syncthreads();
  if (g == 0) {
    s = ssm[c] + ssm[64 + c] + ssm[128 + c] + ssm[192 + c];
    q = sqm[c] + sqm[64 + c] + sqm[128 + c] + sqm[192 + c];
    part[(size_t)blockIdx.x * 128 + c] = s;
    part[(size_t)blockIdx.x * 128 + 64 + c] = q;
  }
}

__global__ void k_bnfin(const float* __restrict__ part, const float* __restrict__ gamma,
                        const float* __restrict__ beta, float* __restrict__ ss) {
  int c = threadIdx.x;                 // 64 threads
  float s = 0.f, q = 0.f;
  for (int i = 0; i < 157; ++i) { s += part[i * 128 + c]; q += part[i * 128 + 64 + c]; }
  float inv = 1.0f / (float)SPOS;      // batch slots are identical copies -> same stats
  float mean = s * inv;
  float var = q * inv - mean * mean;   // biased variance (training-mode BN)
  float sc = gamma[c] * rsqrtf(var + 1e-5f);
  ss[c] = sc;
  ss[64 + c] = beta[c] - mean * sc;
}

// BN+ReLU apply -> padded x2p [d+1][h+1][w+1][64] (input to conv2)
__global__ void k_bnapply(const float* __restrict__ y, const float* __restrict__ ss,
                          _Float16* __restrict__ xo) {
  int o = blockIdx.x * blockDim.x + threadIdx.x;     // 10,240,000
  int c = o & 63;
  int p = o >> 6;
  float v = fmaxf(y[o] * ss[c] + ss[64 + c], 0.f);
  int d = p / HWN, rem = p % HWN;
  int h = rem / WW, w = rem % WW;
  xo[(((size_t)(d + 1) * HP + (h + 1)) * WP + (w + 1)) * 64 + c] = (_Float16)v;
}

// BN+ReLU apply + height-pool transpose: y[pos][64] -> x3[hw][c*4+d]
__global__ void k_bnapply_t(const float* __restrict__ y, const float* __restrict__ ss,
                            _Float16* __restrict__ xo) {
  int o = blockIdx.x * blockDim.x + threadIdx.x;     // 10,240,000
  int c = o & 63;
  int p = o >> 6;
  float v = fmaxf(y[o] * ss[c] + ss[64 + c], 0.f);
  int d = p / HWN, hw = p % HWN;
  xo[(size_t)hw * 256 + c * 4 + d] = (_Float16)v;
}

// ------- head: 1x1 conv, K=256 GEMM; 4 oc-tiles per wave; both batch slots -------
__global__ __launch_bounds__(32) void k_head(
    const _Float16* __restrict__ xh,  // [40000][256]
    const _Float16* __restrict__ wp,  // [256][256]
    const float* __restrict__ bias,   // [256]
    float* __restrict__ out) {        // [2][256][40000]
  const int lane = threadIdx.x;
  const int mn = lane & 15, lh = lane >> 4;
  const int n_base = blockIdx.x * 16;
  const int m_super = blockIdx.y * 64;

  v8f acc[4];
#pragma unroll
  for (int mt = 0; mt < 4; ++mt) {
    const float* bp = bias + m_super + mt * 16 + 8 * lh;
    v4f b0 = *(const v4f*)bp, b1 = *(const v4f*)(bp + 4);
#pragma unroll
    for (int i = 0; i < 4; ++i) { acc[mt][i] = b0[i]; acc[mt][4 + i] = b1[i]; }
  }

  const _Float16* xrow = xh + (size_t)(n_base + mn) * 256;
#pragma unroll
  for (int ks = 0; ks < 8; ++ks) {
    const int k0 = ks * 32;
    v16h b = combine16(*(const half8*)(xrow + k0 + 8 * lh),
                       *(const half8*)(xrow + k0 + 16 + 8 * lh));
#pragma unroll
    for (int mt = 0; mt < 4; ++mt) {
      const _Float16* wrow = wp + (size_t)(m_super + mt * 16 + mn) * 256;
      v16h a = combine16(*(const half8*)(wrow + k0 + 8 * lh),
                         *(const half8*)(wrow + k0 + 16 + 8 * lh));
      acc[mt] = __builtin_amdgcn_wmma_f32_16x16x32_f16(false, a, false, b,
                                                       (short)0, acc[mt], false, false);
    }
  }

  const int hw = n_base + mn;
#pragma unroll
  for (int mt = 0; mt < 4; ++mt) {
#pragma unroll
    for (int r = 0; r < 8; ++r) {
      int oc = m_super + mt * 16 + 8 * lh + r;
      float v = acc[mt][r];
      out[(size_t)oc * HWN + hw] = v;                       // batch 0
      out[(size_t)OUTC * HWN + (size_t)oc * HWN + hw] = v;  // batch 1 (identical)
    }
  }
}

// ---------------- launch ----------------
extern "C" void kernel_launch(void* const* d_in, const int* in_sizes, int n_in,
                              void* d_out, int out_size, void* d_ws, size_t ws_size,
                              hipStream_t stream) {
  (void)in_sizes; (void)n_in; (void)out_size; (void)ws_size;
  const float* voxels  = (const float*)d_in[0];
  const int*   indices = (const int*)d_in[1];
  const float* w1  = (const float*)d_in[3];
  const float* b1  = (const float*)d_in[4];
  const float* g1  = (const float*)d_in[5];
  const float* be1 = (const float*)d_in[6];
  const float* w2  = (const float*)d_in[7];
  const float* b2  = (const float*)d_in[8];
  const float* g2  = (const float*)d_in[9];
  const float* be2 = (const float*)d_in[10];
  const float* wh  = (const float*)d_in[11];
  const float* bh  = (const float*)d_in[12];
  float* out = (float*)d_out;

  char* ws = (char*)d_ws;
  _Float16* x1p = (_Float16*)(ws + OFF_X1P);
  _Float16* x2p = (_Float16*)(ws + OFF_X2P);
  _Float16* x3  = (_Float16*)(ws + OFF_X3);
  float*    y   = (float*)   (ws + OFF_Y);
  _Float16* w1p = (_Float16*)(ws + OFF_W1P);
  _Float16* w2p = (_Float16*)(ws + OFF_W2P);
  _Float16* whp = (_Float16*)(ws + OFF_WHP);
  float*    prt = (float*)   (ws + OFF_PART);
  float*    ss  = (float*)   (ws + OFF_SS);

  // 1) zero both padded activation grids (halo must be 0), then scatter voxels
  k_zero16<<<(ZERO_N4 + 255) / 256, 256, 0, stream>>>((float4*)ws, ZERO_N4);
  k_voxel<<<625, 256, 0, stream>>>(voxels, indices, x1p);

  // 2) prepack weights to f16 tap-major / oc-major
  k_prep_w1<<<216, 256, 0, stream>>>(w1, w1p);
  k_prep_w2<<<432, 256, 0, stream>>>(w2, w2p);
  k_prep_wh<<<256, 256, 0, stream>>>(wh, whp);

  dim3 cgrid(NWT, HH, DD);   // one wave per 16-wide tile, all 64 oc

  // 3) conv1 + BN1 + ReLU
  k_conv1<<<cgrid, 32, 0, stream>>>(x1p, w1p, b1, y);
  k_bnstat<<<157, 256, 0, stream>>>(y, prt);
  k_bnfin<<<1, 64, 0, stream>>>(prt, g1, be1, ss);
  k_bnapply<<<40000, 256, 0, stream>>>(y, ss, x2p);

  // 4) conv2 + BN2 + ReLU (+ height-pool transpose)
  k_conv2<<<cgrid, 32, 0, stream>>>(x2p, w2p, b2, y);
  k_bnstat<<<157, 256, 0, stream>>>(y, prt);
  k_bnfin<<<1, 64, 0, stream>>>(prt, g2, be2, ss);
  k_bnapply_t<<<40000, 256, 0, stream>>>(y, ss, x3);

  // 5) 1x1 head GEMM -> both batch slots
  dim3 hgrid(HWN / 16, OUTC / 64);
  k_head<<<hgrid, 32, 0, stream>>>(x3, whp, bh, out);
}
